// SpatialPatchMoE_68616397521259
// MI455X (gfx1250) — matrix operand order repaired and checked by reading, not compiled, not run on hardware
//
#include <hip/hip_runtime.h>

// ---------------------------------------------------------------------------
// SpatialPatchMoE for gfx1250 (MI455X): bf16 WMMA GEMMs with async-to-LDS
// B-tile staging (CDNA5 GLOBAL_LOAD_ASYNC_TO_LDS + ASYNCcnt), fp32 router.
//   B=16, C=128, H=W=32, PATCH=2 -> N=4096 tokens, d=512
//   E=8 experts, hidden=1024, top-2 gating (dense masked combine)
// K-loop manually unrolled x2: A fragments ping-pong in fixed registers
// (no aCur=aNxt mov chain, no WMMA->VALU hazard NOP window at the handoff).
// ---------------------------------------------------------------------------

typedef __attribute__((ext_vector_type(8)))  __bf16 v8bf;
typedef __attribute__((ext_vector_type(16))) __bf16 v16bf;
typedef __attribute__((ext_vector_type(8)))  float  v8f;

#define N_TOK   4096
#define DIM     512
#define NEXP    8
#define HID     1024
#define LDS_STRIDE_E 40   // elements per LDS B row (32 data + 8 pad) = 80 bytes

__device__ __forceinline__ v8f wmma_bf16(v16bf a, v16bf b, v8f c) {
  return __builtin_amdgcn_wmma_f32_16x16x32_bf16(
      false, a, false, b, (short)0, c, false, false);
}

// Async copy 16B global -> LDS, tracked by ASYNCcnt (no VGPR data).
__device__ __forceinline__ void async_ld_b128(unsigned lds_off, const __bf16* g) {
  asm volatile("global_load_async_to_lds_b128 %0, %1, off"
               :: "v"(lds_off), "v"(g) : "memory");
}
__device__ __forceinline__ void wait_async0() {
  asm volatile("s_wait_asynccnt 0" ::: "memory");
}

// A fragment: row-major A[M,K], lane holds row (m0 + lane&15).
// ISA 16-bit A 16x32 layout: lanes 0-15 K={0..7,16..23}, lanes 16-31 +8.
__device__ __forceinline__ v16bf load_a_frag(const __bf16* __restrict__ A,
                                             int row, int k0, int ld, int hi) {
  const __bf16* p = A + row * ld + k0 + hi * 8;
  v8bf a0 = *(const v8bf*)(p);
  v8bf a1 = *(const v8bf*)(p + 16);
  return __builtin_shufflevector(a0, a1, 0,1,2,3,4,5,6,7,8,9,10,11,12,13,14,15);
}

// B fragment from LDS tile (row = output column, 32 K-contiguous elems/row,
// padded stride). Lane holds col (n-tile + lane&15); K = {0..7,16..23} + 8*(lane>=16).
__device__ __forceinline__ v16bf lds_b_frag(const __bf16* smem, int row, int hi) {
  const __bf16* p = smem + row * LDS_STRIDE_E + hi * 8;
  v8bf b0 = *(const v8bf*)(p);
  v8bf b1 = *(const v8bf*)(p + 16);
  return __builtin_shufflevector(b0, b1, 0,1,2,3,4,5,6,7,8,9,10,11,12,13,14,15);
}

// ---------------------------------------------------------------------------
// Kernel 1: patchify -> bf16 Xp, fp32 router logits, softmax, top-2 gate.
// ---------------------------------------------------------------------------
__global__ void patchify_router(const float* __restrict__ X,   // [16,128,32,32]
                                const float* __restrict__ Rw,  // [512,8]
                                __bf16* __restrict__ Xp,       // [4096,512]
                                float* __restrict__ Gate) {    // [4096,8]
  __shared__ float xs[DIM];
  __shared__ float lg[NEXP];
  const int n  = blockIdx.x;
  const int b  = n >> 8, ph = (n >> 4) & 15, pw = n & 15;
  const int t  = threadIdx.x;

  for (int d = t; d < DIM; d += 256) {
    int c = d >> 2, dy = (d >> 1) & 1, dx = d & 1;
    float v = X[((b * 128 + c) * 32 + (ph * 2 + dy)) * 32 + (pw * 2 + dx)];
    xs[d] = v;
    Xp[n * DIM + d] = (__bf16)v;
  }
  __syncthreads();

  const int wave = t >> 5, lane = t & 31;
  float s = 0.f;
  for (int d = lane; d < DIM; d += 32) s += xs[d] * Rw[d * NEXP + wave];
  #pragma unroll
  for (int off = 16; off > 0; off >>= 1) s += __shfl_down(s, off, 32);
  if (lane == 0) lg[wave] = s;
  __syncthreads();

  if (t == 0) {
    float mx = lg[0];
    #pragma unroll
    for (int e = 1; e < NEXP; ++e) mx = fmaxf(mx, lg[e]);
    float p[NEXP], sum = 0.f;
    #pragma unroll
    for (int e = 0; e < NEXP; ++e) { p[e] = __expf(lg[e] - mx); sum += p[e]; }
    float inv = 1.f / sum;
    #pragma unroll
    for (int e = 0; e < NEXP; ++e) p[e] *= inv;
    int i1 = 0;
    #pragma unroll
    for (int e = 1; e < NEXP; ++e) if (p[e] > p[i1]) i1 = e;   // lowest idx wins ties
    int i2 = (i1 == 0) ? 1 : 0;
    #pragma unroll
    for (int e = 0; e < NEXP; ++e) if (e != i1 && p[e] > p[i2]) i2 = e;
    float denom = 1.f / (p[i1] + p[i2]);
    #pragma unroll
    for (int e = 0; e < NEXP; ++e)
      Gate[n * NEXP + e] = (e == i1 ? p[i1] : (e == i2 ? p[i2] : 0.f)) * denom;
  }
}

// fp32 [K,N] -> bf16 transposed [N,K] (per-expert weight prep, K-contiguous)
__global__ void cvt_transpose(const float* __restrict__ in, __bf16* __restrict__ out,
                              int K, int N) {
  int i = blockIdx.x * 256 + threadIdx.x;   // grid sized exactly K*N/256
  int k = i / N, nn = i % N;
  out[nn * K + k] = (__bf16)in[i];
}

__global__ void zero_f32(float* __restrict__ p, int n) {
  int i = blockIdx.x * 256 + threadIdx.x;
  if (i < n) p[i] = 0.f;
}

// ---------------------------------------------------------------------------
// GEMM1 + SwiGLU. Block = 4 waves = 64(M) x 64(N) of BOTH halves (val+gate).
// B tile (128 rows x 32 K) async-staged in LDS, double buffered.
// ---------------------------------------------------------------------------
__global__ void gemm1_swiglu(const __bf16* __restrict__ Xp,   // [4096,512]
                             const __bf16* __restrict__ W1t,  // [2048,512] N-major
                             __bf16* __restrict__ He) {       // [4096,1024]
  __shared__ __bf16 Bs[2][128 * LDS_STRIDE_E];
  const int t    = threadIdx.x;           // 128 threads
  const int lane = t & 31;
  const int wv   = t >> 5;
  const int hi = lane >> 4, lo = lane & 15;
  const int m0 = blockIdx.x * 64 + wv * 16;
  const int n0 = blockIdx.y * 64;
  const int arow = m0 + lo;
  const unsigned lds0 = (unsigned)(uintptr_t)(void*)&Bs[0][0];

  // issue B tile for K-step `k0` into buffer `buf`: 512 16B chunks, 4/thread
  auto issue_b = [&](int buf, int k0) {
    unsigned base = lds0 + (unsigned)buf * 128u * (LDS_STRIDE_E * 2);
    #pragma unroll
    for (int j = 0; j < 4; ++j) {
      int chunk = t + 128 * j;            // 0..511
      int row   = chunk >> 2;             // 0..127 (LDS row)
      int part  = chunk & 3;              // 16B chunk within row
      int col   = (row < 64) ? (n0 + row) : (HID + n0 + row - 64);
      async_ld_b128(base + (unsigned)(row * (LDS_STRIDE_E * 2) + part * 16),
                    W1t + col * DIM + k0 + part * 8);
    }
  };

  v8f accV[4] = {}; v8f accG[4] = {};

  auto compute = [&](const __bf16* bb, v16bf a) {
    #pragma unroll
    for (int f = 0; f < 4; ++f) {
      v16bf bv = lds_b_frag(bb, f * 16 + lo, hi);
      accV[f] = wmma_bf16(a, bv, accV[f]);
      v16bf bg = lds_b_frag(bb, 64 + f * 16 + lo, hi);
      accG[f] = wmma_bf16(a, bg, accG[f]);
    }
  };

  constexpr int NS = DIM / 32;            // 16 K-steps (even)
  issue_b(0, 0);
  v16bf a0 = load_a_frag(Xp, arow, 0, DIM, hi);
  v16bf a1;

  for (int it = 0; it < NS; it += 2) {
    const int k0 = it * 32;
    // even step: compute buf0, prefetch buf1 (always exists: it+1 < NS)
    wait_async0();
    __syncthreads();
    issue_b(1, k0 + 32);
    a1 = load_a_frag(Xp, arow, k0 + 32, DIM, hi);
    compute(&Bs[0][0], a0);
    // odd step: compute buf1, prefetch buf0 for it+2 if it exists
    wait_async0();
    __syncthreads();
    if (it + 2 < NS) {
      issue_b(0, k0 + 64);
      a0 = load_a_frag(Xp, arow, k0 + 64, DIM, hi);
    }
    compute(&Bs[1][0], a1);
  }

  // C/D layout: element(r, lane) = (M = r + 8*hi, N = lo)
  #pragma unroll
  for (int f = 0; f < 4; ++f) {
    #pragma unroll
    for (int r = 0; r < 8; ++r) {
      float g   = accG[f][r];
      float sig = 1.f / (1.f + __expf(-g));
      int m = m0 + r + hi * 8;
      int nn = n0 + f * 16 + lo;
      He[m * HID + nn] = (__bf16)(accV[f][r] * sig);
    }
  }
}

// ---------------------------------------------------------------------------
// GEMM2 + gated combine + unpatchify: y += gate[m][e] * (He @ w2[e]).
// Same async-LDS pipeline; B tile = 64 rows x 32 K.
// ---------------------------------------------------------------------------
__global__ void gemm2_combine(const __bf16* __restrict__ He,   // [4096,1024]
                              const __bf16* __restrict__ W2t,  // [512,1024] N-major
                              const float* __restrict__ Gate,  // [4096,8]
                              float* __restrict__ Out,         // [16,128,32,32]
                              int expert) {
  __shared__ __bf16 Bs[2][64 * LDS_STRIDE_E];
  const int t    = threadIdx.x;           // 128 threads
  const int lane = t & 31;
  const int wv   = t >> 5;
  const int hi = lane >> 4, lo = lane & 15;
  const int m0 = blockIdx.x * 64 + wv * 16;
  const int n0 = blockIdx.y * 64;
  const int arow = m0 + lo;
  const unsigned lds0 = (unsigned)(uintptr_t)(void*)&Bs[0][0];

  auto issue_b = [&](int buf, int k0) {
    unsigned base = lds0 + (unsigned)buf * 64u * (LDS_STRIDE_E * 2);
    #pragma unroll
    for (int j = 0; j < 2; ++j) {
      int chunk = t + 128 * j;            // 0..255
      int row   = chunk >> 2;             // 0..63
      int part  = chunk & 3;
      async_ld_b128(base + (unsigned)(row * (LDS_STRIDE_E * 2) + part * 16),
                    W2t + (n0 + row) * HID + k0 + part * 8);
    }
  };

  v8f acc[4] = {};

  auto compute = [&](const __bf16* bb, v16bf a) {
    #pragma unroll
    for (int f = 0; f < 4; ++f) {
      v16bf b = lds_b_frag(bb, f * 16 + lo, hi);
      acc[f] = wmma_bf16(a, b, acc[f]);
    }
  };

  constexpr int NS = HID / 32;            // 32 K-steps (even)
  issue_b(0, 0);
  v16bf a0 = load_a_frag(He, arow, 0, HID, hi);
  v16bf a1;

  for (int it = 0; it < NS; it += 2) {
    const int k0 = it * 32;
    wait_async0();
    __syncthreads();
    issue_b(1, k0 + 32);
    a1 = load_a_frag(He, arow, k0 + 32, HID, hi);
    compute(&Bs[0][0], a0);
    wait_async0();
    __syncthreads();
    if (it + 2 < NS) {
      issue_b(0, k0 + 64);
      a0 = load_a_frag(He, arow, k0 + 64, HID, hi);
    }
    compute(&Bs[1][0], a1);
  }

  #pragma unroll
  for (int r = 0; r < 8; ++r) {
    int m = m0 + r + hi * 8;
    float g = Gate[m * NEXP + expert];
    int b = m >> 8, ph = (m >> 4) & 15, pw = m & 15;
    #pragma unroll
    for (int f = 0; f < 4; ++f) {
      int col = n0 + f * 16 + lo;           // d-index in [0,512)
      int c = col >> 2, dy = (col >> 1) & 1, dx = col & 1;
      int idx = ((b * 128 + c) * 32 + (ph * 2 + dy)) * 32 + (pw * 2 + dx);
      Out[idx] += g * acc[f][r];            // experts run sequentially: race-free
    }
  }
}

// ---------------------------------------------------------------------------
// Workspace (~15.2 MB): Xp | Gate | W1t (per-expert) | W2t (per-expert) | He
// ---------------------------------------------------------------------------
extern "C" void kernel_launch(void* const* d_in, const int* in_sizes, int n_in,
                              void* d_out, int out_size, void* d_ws, size_t ws_size,
                              hipStream_t stream) {
  const float* x  = (const float*)d_in[0];   // [16,128,32,32]
  const float* rw = (const float*)d_in[1];   // [512,8]
  const float* w1 = (const float*)d_in[2];   // [8,512,2048]
  const float* w2 = (const float*)d_in[3];   // [8,1024,512]
  float* out = (float*)d_out;                // [16,128,32,32]

  char* ws = (char*)d_ws;
  __bf16* Xp   = (__bf16*)(ws);
  float*  Gate = (float*) (ws + 4194304);
  __bf16* W1t  = (__bf16*)(ws + 4325376);
  __bf16* W2t  = (__bf16*)(ws + 6422528);
  __bf16* He   = (__bf16*)(ws + 7471104);

  patchify_router<<<N_TOK, 256, 0, stream>>>(x, rw, Xp, Gate);
  zero_f32<<<(out_size + 255) / 256, 256, 0, stream>>>(out, out_size);

  for (int e = 0; e < NEXP; ++e) {
    cvt_transpose<<<(DIM * 2 * HID) / 256, 256, 0, stream>>>(
        w1 + (size_t)e * DIM * 2 * HID, W1t, DIM, 2 * HID);   // [512,2048]->[2048,512]
    cvt_transpose<<<(HID * DIM) / 256, 256, 0, stream>>>(
        w2 + (size_t)e * HID * DIM, W2t, HID, DIM);           // [1024,512]->[512,1024]
    gemm1_swiglu<<<dim3(N_TOK / 64, HID / 64), 128, 0, stream>>>(Xp, W1t, He);
    gemm2_combine<<<dim3(N_TOK / 64, DIM / 64), 128, 0, stream>>>(He, W2t, Gate, out, e);
  }
}
